// AttentionLayer_3109556323076
// MI455X (gfx1250) — matrix operand order repaired and checked by reading, not compiled, not run on hardware
//
#include <hip/hip_runtime.h>
#include <stdint.h>

typedef __attribute__((ext_vector_type(16))) __bf16 v16bf;
typedef __attribute__((ext_vector_type(8)))  float  v8f;
typedef __attribute__((ext_vector_type(8)))  __bf16 bf16x8;
typedef __attribute__((ext_vector_type(4)))  __bf16 bf16x4;

#define S_LEN     4096
#define D_DIM     128
#define BM        64        // query rows per workgroup
#define BN        64        // kv rows per iteration
#define NWAVE     4
#define NTHREADS  128
#define QK_STRIDE 136       // bf16 elems/row for Qs,Kb (128 + 8 pad) -> 272B/row
#define V_STRIDE  72        // bf16 elems/row for Vt,P  (64 + 8 pad)  -> 144B/row
#define NBLK      (S_LEN / BN)

#define QS_BYTES  (BM * QK_STRIDE * 2)        // 17408
#define KB_BYTES  (BN * QK_STRIDE * 2)        // 17408
#define VT_BYTES  (D_DIM * V_STRIDE * 2)      // 18432
#define PST_BYTES (NWAVE * 16 * V_STRIDE * 2) // 9216

#define ATTN_SCALE 0.088388347648318447f      // 1/sqrt(128)

// ---------------- fragment loaders ----------------
// A-fragment (16x32 bf16): lane m = l%16, half = l/16.
// elements 0..7  = k0 + half*8 + {0..7};  elements 8..15 = k0 + 16 + half*8 + {0..7}
__device__ __forceinline__ v16bf load_frag_a(const __bf16* base, int row, int stride,
                                             int k0, int half) {
  const __bf16* p = base + row * stride + k0 + half * 8;
  bf16x8 lo = *(const bf16x8*)p;
  bf16x8 hi = *(const bf16x8*)(p + 16);
  v16bf r;
#pragma unroll
  for (int i = 0; i < 8; ++i) { r[i] = lo[i]; r[i + 8] = hi[i]; }
  return r;
}

// B-fragment (32x16 bf16): lane n = l%16; elements 0..7 = k0 + half*8 + {0..7}; 8..15 = k0 + 16 + half*8 + {0..7}.
__device__ __forceinline__ v16bf load_frag_b(const __bf16* base, int row, int stride,
                                             int k0, int half) {
  const __bf16* p = base + row * stride + k0 + half * 8;
  bf16x8 lo = *(const bf16x8*)p;
  bf16x8 hi = *(const bf16x8*)(p + 16);
  v16bf r;
#pragma unroll
  for (int i = 0; i < 8; ++i) { r[i] = lo[i]; r[i + 8] = hi[i]; }
  return r;
}

// ---------------- CDNA5 async copy: global -> LDS (ASYNCcnt) ----------------
__device__ __forceinline__ void async_copy_b128(uint32_t lds_addr, const void* gaddr) {
  asm volatile("global_load_async_to_lds_b128 %0, %1, off"
               :: "v"(lds_addr), "v"(gaddr) : "memory");
}
__device__ __forceinline__ void wait_async_le16() {
  asm volatile("s_wait_asynccnt 0x10" ::: "memory");
}
__device__ __forceinline__ void wait_async_0() {
  asm volatile("s_wait_asynccnt 0x0" ::: "memory");
}

// Issue one KV block: K tile (BN x 128 bf16, contiguous) and Vt tile (128 x BN bf16,
// global row stride S_LEN). 16 async instructions per wave = 8 KB/wave, 32 KB/WG.
__device__ __forceinline__ void issue_tile_copies(uint32_t kb_lds, uint32_t vt_lds,
                                                  const __bf16* Kg, const __bf16* Vtg,
                                                  int kv, int tid) {
#pragma unroll
  for (int j = 0; j < 8; ++j) {              // K: 1024 16B chunks / 128 threads
    int c = tid + j * NTHREADS;
    int row = c >> 4, o = c & 15;            // 16 chunks per 256B row
    async_copy_b128(kb_lds + (uint32_t)(row * QK_STRIDE + o * 8) * 2,
                    Kg + (size_t)(kv + row) * D_DIM + o * 8);
  }
#pragma unroll
  for (int j = 0; j < 8; ++j) {              // Vt: 128 rows x 8 chunks of 16B
    int c = tid + j * NTHREADS;
    int d = c >> 3, o = c & 7;
    async_copy_b128(vt_lds + (uint32_t)(d * V_STRIDE + o * 8) * 2,
                    Vtg + (size_t)d * S_LEN + kv + o * 8);
  }
}

// ---------------- one KV-block of flash attention (shared by both paths) ----
// Fragment loads are software-prefetched 2 steps ahead so each WMMA waits at
// dscnt<=4 instead of dscnt==0 (LDS latency hides under two WMMAs).
__device__ __forceinline__ void attn_block_compute(
    const __bf16* __restrict__ Kt, const __bf16* __restrict__ Vt,
    __bf16* __restrict__ Pw, const v16bf* aq, v8f* oacc,
    float* row_max, float* row_sum, int col, int half) {
  // ---- S = (Q*scale) @ K^T : 16 WMMAs over (nt,kt) ----
  v8f sacc[4];
#pragma unroll
  for (int nt = 0; nt < 4; ++nt)
#pragma unroll
    for (int e = 0; e < 8; ++e) sacc[nt][e] = 0.0f;

  v16bf f0 = load_frag_b(Kt, col, QK_STRIDE, 0, half);   // step 0: nt0,kt0
  v16bf f1 = load_frag_b(Kt, col, QK_STRIDE, 32, half);  // step 1: nt0,kt1
#pragma unroll
  for (int q = 0; q < 16; ++q) {
    const int nt = q >> 2, kt = q & 3;
    v16bf fn = f1;
    if (q + 2 < 16) {
      const int qq = q + 2;
      fn = load_frag_b(Kt, (qq >> 2) * 16 + col, QK_STRIDE, (qq & 3) * 32, half);
    }
    sacc[nt] = __builtin_amdgcn_wmma_f32_16x16x32_bf16(false, aq[kt], false, f0,
                                                       (short)0, sacc[nt], false, false);
    f0 = f1; f1 = fn;
  }

  // ---- online softmax (rows m = r + 8*half; cols span one 16-lane half) ----
#pragma unroll
  for (int r = 0; r < 8; ++r) {
    float t = fmaxf(fmaxf(sacc[0][r], sacc[1][r]), fmaxf(sacc[2][r], sacc[3][r]));
#pragma unroll
    for (int off = 8; off >= 1; off >>= 1)
      t = fmaxf(t, __shfl_xor(t, off, 32));
    float mnew = fmaxf(row_max[r], t);
    float fsc  = __expf(row_max[r] - mnew);
    row_max[r] = mnew;
    float psum = 0.0f;
#pragma unroll
    for (int nt = 0; nt < 4; ++nt) {
      float p = __expf(sacc[nt][r] - mnew);
      sacc[nt][r] = p;
      psum += p;
    }
#pragma unroll
    for (int off = 8; off >= 1; off >>= 1)
      psum += __shfl_xor(psum, off, 32);
    row_sum[r] = row_sum[r] * fsc + psum;
#pragma unroll
    for (int dt = 0; dt < 8; ++dt) oacc[dt][r] *= fsc;
  }

  // ---- P -> per-wave LDS strip (same-wave DS ops are in order) ----
#pragma unroll
  for (int nt = 0; nt < 4; ++nt)
#pragma unroll
    for (int r = 0; r < 8; ++r)
      Pw[(r + 8 * half) * V_STRIDE + nt * 16 + col] = (__bf16)sacc[nt][r];

  // ---- O += P @ V : 16 WMMAs over (dt,kt), prefetched the same way ----
  v16bf ap0 = load_frag_a(Pw, col, V_STRIDE, 0, half);
  v16bf ap1 = load_frag_a(Pw, col, V_STRIDE, 32, half);
  v16bf g0 = load_frag_b(Vt, col, V_STRIDE, 0, half);    // step 0: dt0,kt0
  v16bf g1 = load_frag_b(Vt, col, V_STRIDE, 32, half);   // step 1: dt0,kt1
#pragma unroll
  for (int p = 0; p < 16; ++p) {
    const int dt = p >> 1, kt = p & 1;
    v16bf gn = g1;
    if (p + 2 < 16) {
      const int pp = p + 2;
      gn = load_frag_b(Vt, (pp >> 1) * 16 + col, V_STRIDE, (pp & 1) * 32, half);
    }
    oacc[dt] = __builtin_amdgcn_wmma_f32_16x16x32_bf16(false, kt ? ap1 : ap0, false, g0,
                                                       (short)0, oacc[dt], false, false);
    g0 = g1; g1 = gn;
  }
}

// ---------------- main attention kernel ----------------
template <bool PRECONV>
__global__ __launch_bounds__(NTHREADS)
void flash_attn_bf16_wmma(const float* __restrict__ Qf, const float* __restrict__ Kf,
                          const float* __restrict__ Vf,
                          const __bf16* __restrict__ Qb, const __bf16* __restrict__ Kbg,
                          const __bf16* __restrict__ Vtg,
                          float* __restrict__ O) {
  constexpr int LDS_BYTES = PRECONV ? (2 * KB_BYTES + 2 * VT_BYTES + PST_BYTES)
                                    : (QS_BYTES + KB_BYTES + VT_BYTES + PST_BYTES);
  __shared__ __attribute__((aligned(16))) unsigned char smem[LDS_BYTES];

  const int tid  = threadIdx.x;
  const int wave = tid >> 5;
  const int lane = tid & 31;
  const int half = lane >> 4;
  const int col  = lane & 15;

  const int q0 = blockIdx.x * BM;
  const int b  = blockIdx.y;

  float* Og = O + (size_t)b * S_LEN * D_DIM;

  v16bf aq[4];
  v8f oacc[8];
#pragma unroll
  for (int dt = 0; dt < 8; ++dt)
#pragma unroll
    for (int e = 0; e < 8; ++e) oacc[dt][e] = 0.0f;
  float row_max[8], row_sum[8];
#pragma unroll
  for (int r = 0; r < 8; ++r) { row_max[r] = -1e30f; row_sum[r] = 0.0f; }

  if constexpr (PRECONV) {
    // ---- fast path: bf16 operands pre-staged in global workspace ----
    uint32_t lds_base = (uint32_t)(uintptr_t)(void*)&smem[0];
    __bf16* Pw = (__bf16*)(smem + 2 * KB_BYTES + 2 * VT_BYTES) + wave * 16 * V_STRIDE;

    const __bf16* Qgb = Qb + ((size_t)b * S_LEN + q0) * D_DIM;
    const __bf16* Kgb = Kbg + (size_t)b * S_LEN * D_DIM;
    const __bf16* Vgb = Vtg + (size_t)b * D_DIM * S_LEN;

#pragma unroll
    for (int kt = 0; kt < 4; ++kt)
      aq[kt] = load_frag_a(Qgb, wave * 16 + col, D_DIM, kt * 32, half);

    issue_tile_copies(lds_base, lds_base + 2 * KB_BYTES, Kgb, Vgb, 0, tid);

    for (int i = 0; i < NBLK; ++i) {
      const int cur = i & 1;
      const int nxt = cur ^ 1;
      if (i + 1 < NBLK) {
        issue_tile_copies(lds_base + (uint32_t)(nxt * KB_BYTES),
                          lds_base + (uint32_t)(2 * KB_BYTES + nxt * VT_BYTES),
                          Kgb, Vgb, (i + 1) * BN, tid);
        wait_async_le16();          // oldest 16 (block i) complete; async done in order
      } else {
        wait_async_0();
      }
      __syncthreads();              // every wave's slice of block i is in LDS

      const __bf16* Kt = (const __bf16*)(smem + cur * KB_BYTES);
      const __bf16* Vt = (const __bf16*)(smem + 2 * KB_BYTES + cur * VT_BYTES);
      attn_block_compute(Kt, Vt, Pw, aq, oacc, row_max, row_sum, col, half);

      __syncthreads();              // block i buffers free for the i+2 prefetch
    }
  } else {
    // ---- fallback path: self-contained, converts fp32 in-kernel ----
    __bf16* Qs = (__bf16*)(smem + 0);
    __bf16* Kt = (__bf16*)(smem + QS_BYTES);
    __bf16* Vt = (__bf16*)(smem + QS_BYTES + KB_BYTES);
    __bf16* Pw = (__bf16*)(smem + QS_BYTES + KB_BYTES + VT_BYTES) + wave * 16 * V_STRIDE;

    const float* Qg = Qf + (size_t)b * S_LEN * D_DIM;
    const float* Kg = Kf + (size_t)b * S_LEN * D_DIM;
    const float* Vg = Vf + (size_t)b * S_LEN * D_DIM;

    {
      const float4* Qv = (const float4*)(Qg + (size_t)q0 * D_DIM);
#pragma unroll
      for (int i = 0; i < 16; ++i) {
        int idx = tid + i * NTHREADS;
        int row = idx >> 5, c4 = (idx & 31) << 2;
        float4 q = Qv[idx];
        bf16x4 h;
        h[0] = (__bf16)(q.x * ATTN_SCALE); h[1] = (__bf16)(q.y * ATTN_SCALE);
        h[2] = (__bf16)(q.z * ATTN_SCALE); h[3] = (__bf16)(q.w * ATTN_SCALE);
        *(bf16x4*)&Qs[row * QK_STRIDE + c4] = h;
      }
    }
    __syncthreads();
#pragma unroll
    for (int kt = 0; kt < 4; ++kt)
      aq[kt] = load_frag_a(Qs, wave * 16 + col, QK_STRIDE, kt * 32, half);

    for (int kv = 0; kv < S_LEN; kv += BN) {
      __syncthreads();
      const float4* Kv = (const float4*)(Kg + (size_t)kv * D_DIM);
#pragma unroll
      for (int i = 0; i < 16; ++i) {
        int idx = tid + i * NTHREADS;
        int row = idx >> 5, c4 = (idx & 31) << 2;
        float4 k4 = Kv[idx];
        bf16x4 h;
        h[0] = (__bf16)k4.x; h[1] = (__bf16)k4.y; h[2] = (__bf16)k4.z; h[3] = (__bf16)k4.w;
        *(bf16x4*)&Kt[row * QK_STRIDE + c4] = h;
      }
      const float4* Vv = (const float4*)(Vg + (size_t)kv * D_DIM);
#pragma unroll
      for (int i = 0; i < 16; ++i) {
        int idx = tid + i * NTHREADS;
        int n = idx >> 5, c4 = (idx & 31) << 2;
        float4 v4 = Vv[idx];
        Vt[(c4 + 0) * V_STRIDE + n] = (__bf16)v4.x;
        Vt[(c4 + 1) * V_STRIDE + n] = (__bf16)v4.y;
        Vt[(c4 + 2) * V_STRIDE + n] = (__bf16)v4.z;
        Vt[(c4 + 3) * V_STRIDE + n] = (__bf16)v4.w;
      }
      __syncthreads();

      attn_block_compute(Kt, Vt, Pw, aq, oacc, row_max, row_sum, col, half);
    }
  }

  // ---- normalize and write out ----
#pragma unroll
  for (int r = 0; r < 8; ++r) {
    float inv = 1.0f / row_sum[r];
    int qrow = q0 + wave * 16 + r + 8 * half;
#pragma unroll
    for (int dt = 0; dt < 8; ++dt)
      Og[(size_t)qrow * D_DIM + dt * 16 + col] = oacc[dt][r] * inv;
  }
}

// ---------------- pre-pass kernels ----------------
__global__ __launch_bounds__(256)
void cvt_bf16_kernel(const float* __restrict__ in, __bf16* __restrict__ out,
                     float scale, long n4) {
  long i = (long)blockIdx.x * blockDim.x + threadIdx.x;
  if (i >= n4) return;
  float4 v = ((const float4*)in)[i];
  bf16x4 h;
  h[0] = (__bf16)(v.x * scale); h[1] = (__bf16)(v.y * scale);
  h[2] = (__bf16)(v.z * scale); h[3] = (__bf16)(v.w * scale);
  ((bf16x4*)out)[i] = h;
}

// V[b][n][d] fp32 -> Vt[b][d][n] bf16, 64x64 tiles through LDS
__global__ __launch_bounds__(256)
void transpose_v_kernel(const float* __restrict__ V, __bf16* __restrict__ Vt) {
  __shared__ __bf16 tile[64][72];
  const int b = blockIdx.z, n0 = blockIdx.x * 64, d0 = blockIdx.y * 64;
  const float* Vb = V + (size_t)b * S_LEN * D_DIM;
  __bf16* Vtb = Vt + (size_t)b * D_DIM * S_LEN;
  const int tid = threadIdx.x;
#pragma unroll
  for (int j = 0; j < 4; ++j) {           // 1024 float4 reads
    int c = tid + j * 256;
    int r = c >> 4, o = (c & 15) << 2;
    float4 v = *(const float4*)(Vb + (size_t)(n0 + r) * D_DIM + d0 + o);
    tile[r][o + 0] = (__bf16)v.x; tile[r][o + 1] = (__bf16)v.y;
    tile[r][o + 2] = (__bf16)v.z; tile[r][o + 3] = (__bf16)v.w;
  }
  __syncthreads();
#pragma unroll
  for (int j = 0; j < 2; ++j) {           // 512 bf16x8 writes
    int c = tid + j * 256;
    int d = c >> 3, o = (c & 7) << 3;
    bf16x8 h;
#pragma unroll
    for (int e = 0; e < 8; ++e) h[e] = tile[o + e][d];
    *(bf16x8*)(Vtb + (size_t)(d0 + d) * S_LEN + n0 + o) = h;
  }
}

// ---------------- launcher ----------------
extern "C" void kernel_launch(void* const* d_in, const int* in_sizes, int n_in,
                              void* d_out, int out_size, void* d_ws, size_t ws_size,
                              hipStream_t stream) {
  const float* Q = (const float*)d_in[0];
  const float* K = (const float*)d_in[1];
  const float* V = (const float*)d_in[2];
  float*       O = (float*)d_out;

  const int batches = in_sizes[0] / (S_LEN * D_DIM);
  const size_t per  = (size_t)batches * S_LEN * D_DIM;
  const size_t need = 3 * per * sizeof(__bf16);

  dim3 agrid(S_LEN / BM, batches);
  dim3 ablock(NTHREADS);

  if (d_ws && ws_size >= need) {
    __bf16* Qb  = (__bf16*)d_ws;
    __bf16* Kbg = Qb + per;
    __bf16* Vtg = Kbg + per;
    long n4 = (long)(per / 4);
    int cvb = (int)((n4 + 255) / 256);
    cvt_bf16_kernel<<<cvb, 256, 0, stream>>>(Q, Qb, ATTN_SCALE, n4);
    cvt_bf16_kernel<<<cvb, 256, 0, stream>>>(K, Kbg, 1.0f, n4);
    dim3 tgrid(S_LEN / 64, D_DIM / 64, batches);
    transpose_v_kernel<<<tgrid, 256, 0, stream>>>(V, Vtg);
    flash_attn_bf16_wmma<true><<<agrid, ablock, 0, stream>>>(
        Q, K, V, Qb, Kbg, Vtg, O);
  } else {
    flash_attn_bf16_wmma<false><<<agrid, ablock, 0, stream>>>(
        Q, K, V, nullptr, nullptr, nullptr, O);
  }
}
